// HareMoE_56667798504234
// MI455X (gfx1250) — matrix operand order repaired and checked by reading, not compiled, not run on hardware
//
#include <hip/hip_runtime.h>

#define HID  2048
#define FFN  4096
#define NTOK 4096
#define NEXP 8
#define LDSP 40   // padded row stride (ushorts): 64B row + 16B pad -> conflict-free b128 reads

typedef __attribute__((ext_vector_type(16))) __bf16        v16bf;
typedef __attribute__((ext_vector_type(8)))  float         v8f;
typedef __attribute__((ext_vector_type(4)))  float         f32x4;
typedef __attribute__((ext_vector_type(4)))  unsigned int  u32x4;
typedef __attribute__((ext_vector_type(8)))  unsigned int  u32x8;
typedef __attribute__((ext_vector_type(2)))  unsigned int  u32x2;
typedef __attribute__((ext_vector_type(8)))  int           i32x8;
typedef __attribute__((ext_vector_type(4)))  int           i32x4;

#if __has_builtin(__builtin_amdgcn_tensor_load_to_lds) && \
    __has_builtin(__builtin_amdgcn_s_wait_tensorcnt)
#define USE_TDM 1
#else
#define USE_TDM 0
#endif

// Pack two floats into packed bf16 (backend fuses into v_cvt_pk_bf16_f32).
static __device__ __forceinline__ unsigned int pack_bf16(float a, float b) {
  __bf16 ha = (__bf16)a, hb = (__bf16)b;
  unsigned short ua = __builtin_bit_cast(unsigned short, ha);
  unsigned short ub = __builtin_bit_cast(unsigned short, hb);
  return (unsigned)ua | ((unsigned)ub << 16);
}

static __device__ __forceinline__ unsigned short f2bf(float f) {
  __bf16 h = (__bf16)f;
  return __builtin_bit_cast(unsigned short, h);
}

// Build a 16xbf16 WMMA fragment from two 16B LDS reads (ds_load_b128 each).
static __device__ __forceinline__ v16bf ld_frag(const unsigned short* p, int o0, int o1) {
  u32x4 a = *(const u32x4*)(p + o0);
  u32x4 b = *(const u32x4*)(p + o1);
  u32x8 w;
  w[0]=a[0]; w[1]=a[1]; w[2]=a[2]; w[3]=a[3];
  w[4]=b[0]; w[5]=b[1]; w[6]=b[2]; w[7]=b[3];
  return __builtin_bit_cast(v16bf, w);
}

static __device__ __forceinline__ v8f wmma_bf16(v16bf a, v16bf b, v8f c) {
  return __builtin_amdgcn_wmma_f32_16x16x32_bf16(false, a, false, b, (short)0, c, false, false);
}

static __device__ __forceinline__ float fast_silu_mul(float g, float u) {
  float s = __builtin_amdgcn_rcpf(1.f + __expf(-g));
  return g * s * u;
}

#if USE_TDM
// Issue a TDM load of a 128-row x 32-col bf16 tile (row stride FFN elements)
// from global `gtile` into LDS offset `lds_off`. D# pad feature inserts 16B
// after every 64B row -> matches the LDSP=40 padded LDS layout.
static __device__ __forceinline__ void tdm_load_tile(const unsigned short* gtile,
                                                     unsigned lds_off) {
  unsigned long long ga = (unsigned long long)(size_t)gtile;
  u32x4 g0;
  g0[0] = 1u;                                          // count=1 (user descriptor)
  g0[1] = lds_off;                                     // lds_addr (bytes)
  g0[2] = (unsigned)(ga & 0xFFFFFFFFu);                // global_addr[31:0]
  g0[3] = (unsigned)((ga >> 32) & 0x01FFFFFFu) | (2u << 30);  // addr[56:32] | type=2
  const unsigned TD = 1u << 30;                        // huge tensor dims: no OOB clip
  i32x8 g1;
  g1[0] = (int)((1u << 16)      // data_size = 2 bytes
              | (1u << 20)      // pad_enable
              | (3u << 22)      // pad_interval: every 8x8B = 64B
              | (3u << 25));    // pad_amount: 4 DWORDs = 16B
  g1[1] = (int)((TD & 0xFFFFu) << 16);                 // tensor_dim0[15:0]
  g1[2] = (int)((TD >> 16) | ((TD & 0xFFFFu) << 16));  // dim0[31:16] | dim1[15:0]
  g1[3] = (int)((TD >> 16) | (32u << 16));             // dim1[31:16] | tile_dim0=32
  g1[4] = (int)128u;                                   // tile_dim1=128, tile_dim2=0
  g1[5] = (int)FFN;                                    // tensor_dim0_stride[31:0]
  g1[6] = 0;                                           // stride0[47:32] | stride1[15:0]
  g1[7] = 0;                                           // stride1[47:16]
  i32x4 z4 = {0, 0, 0, 0};
  i32x8 z8 = {0, 0, 0, 0, 0, 0, 0, 0};
  // 6-arg variant (clang-23 / therock-10 headers): (g0, g1, g2, g3, g4, cpol)
  __builtin_amdgcn_tensor_load_to_lds(g0, g1, z4, z4, z8, 0);
}
#endif

// ---------------- routing: one wave32 per token ----------------
__global__ __launch_bounds__(256)
void moe_router(const float* __restrict__ x, const float* __restrict__ gw,
                float* __restrict__ cw) {
  const int wave = threadIdx.x >> 5;
  const int lane = threadIdx.x & 31;
  const int t = blockIdx.x * 8 + wave;
  float acc[NEXP];
  #pragma unroll
  for (int e = 0; e < NEXP; ++e) acc[e] = 0.f;
  const float* xr = x + (size_t)t * HID;
  for (int k = lane; k < HID; k += 32) {
    float xv = xr[k];
    #pragma unroll
    for (int e = 0; e < NEXP; ++e) acc[e] += xv * gw[e * HID + k];
  }
  #pragma unroll
  for (int off = 16; off > 0; off >>= 1) {
    #pragma unroll
    for (int e = 0; e < NEXP; ++e) acc[e] += __shfl_xor(acc[e], off, 32);
  }
  float m1 = -3.4e38f; int i1 = 0;
  #pragma unroll
  for (int e = 0; e < NEXP; ++e) if (acc[e] > m1) { m1 = acc[e]; i1 = e; }
  float m2 = -3.4e38f; int i2 = 0;
  #pragma unroll
  for (int e = 0; e < NEXP; ++e) if (e != i1 && acc[e] > m2) { m2 = acc[e]; i2 = e; }
  float p2 = __expf(m2 - m1);
  float wa = __builtin_amdgcn_rcpf(1.f + p2);
  float wb = p2 * wa;
  if (lane < NEXP) {
    float v = (lane == i1) ? wa : ((lane == i2) ? wb : 0.f);
    cw[(size_t)t * NEXP + lane] = v;
  }
}

__global__ __launch_bounds__(256)
void zero_out(float* __restrict__ p, int n4) {
  int i = blockIdx.x * blockDim.x + threadIdx.x;
  f32x4 z = {0.f, 0.f, 0.f, 0.f};
  if (i < n4) ((f32x4*)p)[i] = z;
}

// ------------- GEMM1: hb = bf16( silu(x@w1^T) * (x@w3^T) ) for one expert -------------
// block 128(M) x 64(N), BK=32; 8 waves 4x2, wave tile 32x32; double-buffered LDS.
__global__ __launch_bounds__(256)
void moe_gemm1(const float* __restrict__ x, const float* __restrict__ w1e,
               const float* __restrict__ w3e, unsigned short* __restrict__ hb) {
  __shared__ unsigned short sA [2][128 * LDSP];
  __shared__ unsigned short sB1[2][ 64 * LDSP];
  __shared__ unsigned short sB3[2][ 64 * LDSP];

  const int tid  = threadIdx.x;
  const int lane = tid & 31;
  const int wave = tid >> 5;
  const int wm   = wave & 3;
  const int wn   = wave >> 2;
  const int half = (lane < 16) ? 0 : 1;
  const int lrow = lane & 15;
  const int mbase = blockIdx.y * 128;
  const int nbase = blockIdx.x * 64;
  const int NK = HID / 32;

  v8f zf = {0.f,0.f,0.f,0.f,0.f,0.f,0.f,0.f};
  v8f accg[2][2], accu[2][2];
  #pragma unroll
  for (int i = 0; i < 2; ++i)
    #pragma unroll
    for (int j = 0; j < 2; ++j) { accg[i][j] = zf; accu[i][j] = zf; }

  int rowA[4], c4A[4], rowB[2], c4B[2];
  #pragma unroll
  for (int i = 0; i < 4; ++i) { int c = tid + i * 256; rowA[i] = c >> 3; c4A[i] = c & 7; }
  #pragma unroll
  for (int i = 0; i < 2; ++i) { int c = tid + i * 256; rowB[i] = c >> 3; c4B[i] = c & 7; }

  f32x4 ra[4], rb1[2], rb3[2];

  #pragma unroll
  for (int i = 0; i < 4; ++i)
    ra[i] = *(const f32x4*)(x + (size_t)(mbase + rowA[i]) * HID + c4A[i] * 4);
  #pragma unroll
  for (int i = 0; i < 2; ++i) {
    size_t off = (size_t)(nbase + rowB[i]) * HID + c4B[i] * 4;
    rb1[i] = *(const f32x4*)(w1e + off);
    rb3[i] = *(const f32x4*)(w3e + off);
  }
  #pragma unroll
  for (int i = 0; i < 4; ++i) {
    u32x2 p; p[0] = pack_bf16(ra[i][0], ra[i][1]); p[1] = pack_bf16(ra[i][2], ra[i][3]);
    *(u32x2*)(sA[0] + rowA[i] * LDSP + c4A[i] * 4) = p;
  }
  #pragma unroll
  for (int i = 0; i < 2; ++i) {
    u32x2 p; p[0] = pack_bf16(rb1[i][0], rb1[i][1]); p[1] = pack_bf16(rb1[i][2], rb1[i][3]);
    *(u32x2*)(sB1[0] + rowB[i] * LDSP + c4B[i] * 4) = p;
    p[0] = pack_bf16(rb3[i][0], rb3[i][1]); p[1] = pack_bf16(rb3[i][2], rb3[i][3]);
    *(u32x2*)(sB3[0] + rowB[i] * LDSP + c4B[i] * 4) = p;
  }
  __syncthreads();

  const int oA = (wm * 32 + lrow) * LDSP + (half ? 8 : 0);
  const int oB = (wn * 32 + lrow) * LDSP + (half ? 8 : 0);

  for (int kt = 0; kt < NK - 1; ++kt) {
    const int cur = kt & 1, nxt = cur ^ 1;
    const int kof = (kt + 1) * 32;
    #pragma unroll
    for (int i = 0; i < 4; ++i)
      ra[i] = *(const f32x4*)(x + (size_t)(mbase + rowA[i]) * HID + kof + c4A[i] * 4);
    #pragma unroll
    for (int i = 0; i < 2; ++i) {
      size_t off = (size_t)(nbase + rowB[i]) * HID + kof + c4B[i] * 4;
      rb1[i] = *(const f32x4*)(w1e + off);
      rb3[i] = *(const f32x4*)(w3e + off);
    }
    v16bf af[2], b1f[2], b3f[2];
    #pragma unroll
    for (int i = 0; i < 2; ++i) af[i]  = ld_frag(sA[cur],  oA + i * 16 * LDSP, oA + i * 16 * LDSP + 16);
    #pragma unroll
    for (int j = 0; j < 2; ++j) {
      b1f[j] = ld_frag(sB1[cur], oB + j * 16 * LDSP, oB + j * 16 * LDSP + 16);
      b3f[j] = ld_frag(sB3[cur], oB + j * 16 * LDSP, oB + j * 16 * LDSP + 16);
    }
    #pragma unroll
    for (int i = 0; i < 2; ++i)
      #pragma unroll
      for (int j = 0; j < 2; ++j) {
        accg[i][j] = wmma_bf16(af[i], b1f[j], accg[i][j]);
        accu[i][j] = wmma_bf16(af[i], b3f[j], accu[i][j]);
      }
    #pragma unroll
    for (int i = 0; i < 4; ++i) {
      u32x2 p; p[0] = pack_bf16(ra[i][0], ra[i][1]); p[1] = pack_bf16(ra[i][2], ra[i][3]);
      *(u32x2*)(sA[nxt] + rowA[i] * LDSP + c4A[i] * 4) = p;
    }
    #pragma unroll
    for (int i = 0; i < 2; ++i) {
      u32x2 p; p[0] = pack_bf16(rb1[i][0], rb1[i][1]); p[1] = pack_bf16(rb1[i][2], rb1[i][3]);
      *(u32x2*)(sB1[nxt] + rowB[i] * LDSP + c4B[i] * 4) = p;
      p[0] = pack_bf16(rb3[i][0], rb3[i][1]); p[1] = pack_bf16(rb3[i][2], rb3[i][3]);
      *(u32x2*)(sB3[nxt] + rowB[i] * LDSP + c4B[i] * 4) = p;
    }
    __syncthreads();
  }

  {
    const int cur = (NK - 1) & 1;
    v16bf af[2], b1f[2], b3f[2];
    #pragma unroll
    for (int i = 0; i < 2; ++i) af[i]  = ld_frag(sA[cur],  oA + i * 16 * LDSP, oA + i * 16 * LDSP + 16);
    #pragma unroll
    for (int j = 0; j < 2; ++j) {
      b1f[j] = ld_frag(sB1[cur], oB + j * 16 * LDSP, oB + j * 16 * LDSP + 16);
      b3f[j] = ld_frag(sB3[cur], oB + j * 16 * LDSP, oB + j * 16 * LDSP + 16);
    }
    #pragma unroll
    for (int i = 0; i < 2; ++i)
      #pragma unroll
      for (int j = 0; j < 2; ++j) {
        accg[i][j] = wmma_bf16(af[i], b1f[j], accg[i][j]);
        accu[i][j] = wmma_bf16(af[i], b3f[j], accu[i][j]);
      }
  }

  #pragma unroll
  for (int i = 0; i < 2; ++i) {
    int rowb = mbase + wm * 32 + i * 16 + (half ? 8 : 0);
    #pragma unroll
    for (int j = 0; j < 2; ++j) {
      int col = nbase + wn * 32 + j * 16 + lrow;
      #pragma unroll
      for (int r = 0; r < 8; ++r) {
        float h = fast_silu_mul(accg[i][j][r], accu[i][j][r]);
        hb[(size_t)(rowb + r) * FFN + col] = f2bf(h);
      }
    }
  }
}

// ------------- GEMM2: out += cw[:,e] * (hb @ w2^T) -------------
// A tile (bf16 hb) fetched by the Tensor Data Mover straight into padded LDS;
// B tile (fp32 w2) staged through VGPRs with on-the-fly bf16 conversion.
__global__ __launch_bounds__(256)
void moe_gemm2(const unsigned short* __restrict__ hb, const float* __restrict__ w2e,
               const float* __restrict__ cw, int e, float* __restrict__ out) {
  __shared__ unsigned short sA[2][128 * LDSP];
  __shared__ unsigned short sB[2][ 64 * LDSP];

  const int tid  = threadIdx.x;
  const int lane = tid & 31;
  const int wave = tid >> 5;
  const int wm   = wave & 3;
  const int wn   = wave >> 2;
  const int half = (lane < 16) ? 0 : 1;
  const int lrow = lane & 15;
  const int mbase = blockIdx.y * 128;
  const int nbase = blockIdx.x * 64;
  const int NK = FFN / 32;

  v8f zf = {0.f,0.f,0.f,0.f,0.f,0.f,0.f,0.f};
  v8f acc[2][2];
  #pragma unroll
  for (int i = 0; i < 2; ++i)
    #pragma unroll
    for (int j = 0; j < 2; ++j) acc[i][j] = zf;

  int rowB[2], c4B[2];
  #pragma unroll
  for (int i = 0; i < 2; ++i) { int c = tid + i * 256; rowB[i] = c >> 3; c4B[i] = c & 7; }
#if !USE_TDM
  int rowA[2], c8A[2];
  #pragma unroll
  for (int i = 0; i < 2; ++i) { int c = tid + i * 256; rowA[i] = c >> 2; c8A[i] = c & 3; }
  u32x4 ra[2];
#endif
  f32x4 rb[2];

  const unsigned short* gA = hb + (size_t)mbase * FFN;
#if USE_TDM
  const unsigned ldsA[2] = { (unsigned)(size_t)&sA[0][0], (unsigned)(size_t)&sA[1][0] };
#endif

  // ---- prologue: K-tile 0 -> buffer 0
#if USE_TDM
  if (wave == 0) tdm_load_tile(gA, ldsA[0]);
#else
  #pragma unroll
  for (int i = 0; i < 2; ++i)
    ra[i] = *(const u32x4*)(gA + (size_t)rowA[i] * FFN + c8A[i] * 8);
#endif
  #pragma unroll
  for (int i = 0; i < 2; ++i)
    rb[i] = *(const f32x4*)(w2e + (size_t)(nbase + rowB[i]) * FFN + c4B[i] * 4);
#if !USE_TDM
  #pragma unroll
  for (int i = 0; i < 2; ++i)
    *(u32x4*)(sA[0] + rowA[i] * LDSP + c8A[i] * 8) = ra[i];
#endif
  #pragma unroll
  for (int i = 0; i < 2; ++i) {
    u32x2 p; p[0] = pack_bf16(rb[i][0], rb[i][1]); p[1] = pack_bf16(rb[i][2], rb[i][3]);
    *(u32x2*)(sB[0] + rowB[i] * LDSP + c4B[i] * 4) = p;
  }
#if USE_TDM
  __builtin_amdgcn_s_wait_tensorcnt(0);
#endif
  __syncthreads();

  const int oA = (wm * 32 + lrow) * LDSP + (half ? 8 : 0);
  const int oB = (wn * 32 + lrow) * LDSP + (half ? 8 : 0);

  for (int kt = 0; kt < NK - 1; ++kt) {
    const int cur = kt & 1, nxt = cur ^ 1;
    const int kof = (kt + 1) * 32;
    // next A tile: TDM DMA straight into the other LDS buffer (overlaps WMMAs)
#if USE_TDM
    if (wave == 0) tdm_load_tile(gA + kof, ldsA[nxt]);
#else
    #pragma unroll
    for (int i = 0; i < 2; ++i)
      ra[i] = *(const u32x4*)(gA + (size_t)rowA[i] * FFN + kof + c8A[i] * 8);
#endif
    #pragma unroll
    for (int i = 0; i < 2; ++i)
      rb[i] = *(const f32x4*)(w2e + (size_t)(nbase + rowB[i]) * FFN + kof + c4B[i] * 4);

    v16bf af[2], bf[2];
    #pragma unroll
    for (int i = 0; i < 2; ++i) af[i] = ld_frag(sA[cur], oA + i * 16 * LDSP, oA + i * 16 * LDSP + 16);
    #pragma unroll
    for (int j = 0; j < 2; ++j) bf[j] = ld_frag(sB[cur], oB + j * 16 * LDSP, oB + j * 16 * LDSP + 16);
    #pragma unroll
    for (int i = 0; i < 2; ++i)
      #pragma unroll
      for (int j = 0; j < 2; ++j)
        acc[i][j] = wmma_bf16(af[i], bf[j], acc[i][j]);

#if !USE_TDM
    #pragma unroll
    for (int i = 0; i < 2; ++i)
      *(u32x4*)(sA[nxt] + rowA[i] * LDSP + c8A[i] * 8) = ra[i];
#endif
    #pragma unroll
    for (int i = 0; i < 2; ++i) {
      u32x2 p; p[0] = pack_bf16(rb[i][0], rb[i][1]); p[1] = pack_bf16(rb[i][2], rb[i][3]);
      *(u32x2*)(sB[nxt] + rowB[i] * LDSP + c4B[i] * 4) = p;
    }
#if USE_TDM
    __builtin_amdgcn_s_wait_tensorcnt(0);
#endif
    __syncthreads();
  }

  {
    const int cur = (NK - 1) & 1;
    v16bf af[2], bf[2];
    #pragma unroll
    for (int i = 0; i < 2; ++i) af[i] = ld_frag(sA[cur], oA + i * 16 * LDSP, oA + i * 16 * LDSP + 16);
    #pragma unroll
    for (int j = 0; j < 2; ++j) bf[j] = ld_frag(sB[cur], oB + j * 16 * LDSP, oB + j * 16 * LDSP + 16);
    #pragma unroll
    for (int i = 0; i < 2; ++i)
      #pragma unroll
      for (int j = 0; j < 2; ++j)
        acc[i][j] = wmma_bf16(af[i], bf[j], acc[i][j]);
  }

  // epilogue: out[t, col] += cw[t, e] * y   (kernels per expert are stream-serialized)
  #pragma unroll
  for (int i = 0; i < 2; ++i) {
    int rowb = mbase + wm * 32 + i * 16 + (half ? 8 : 0);
    float cr[8];
    #pragma unroll
    for (int r = 0; r < 8; ++r) cr[r] = cw[(size_t)(rowb + r) * NEXP + e];
    #pragma unroll
    for (int j = 0; j < 2; ++j) {
      int col = nbase + wn * 32 + j * 16 + lrow;
      #pragma unroll
      for (int r = 0; r < 8; ++r) {
        float* po = out + (size_t)(rowb + r) * HID + col;
        *po += cr[r] * acc[i][j][r];
      }
    }
  }
}

extern "C" void kernel_launch(void* const* d_in, const int* in_sizes, int n_in,
                              void* d_out, int out_size, void* d_ws, size_t ws_size,
                              hipStream_t stream) {
  const float* x  = (const float*)d_in[0];   // [2,2048,2048] -> [4096,2048]
  const float* gw = (const float*)d_in[1];   // [8,2048]
  const float* w1 = (const float*)d_in[2];   // [8,4096,2048]
  const float* w2 = (const float*)d_in[3];   // [8,2048,4096]
  const float* w3 = (const float*)d_in[4];   // [8,4096,2048]
  float* out = (float*)d_out;

  float* cw = (float*)d_ws;                                           // [4096,8] f32 = 128 KB
  unsigned short* hb = (unsigned short*)((char*)d_ws +
                       (size_t)NTOK * NEXP * sizeof(float));          // [4096,4096] bf16 = 32 MB

  moe_router<<<NTOK / 8, 256, 0, stream>>>(x, gw, cw);
  zero_out<<<(NTOK * HID / 4 + 255) / 256, 256, 0, stream>>>(out, NTOK * HID / 4);

  for (int e = 0; e < NEXP; ++e) {
    moe_gemm1<<<dim3(FFN / 64, NTOK / 128), 256, 0, stream>>>(
        x, w1 + (size_t)e * FFN * HID, w3 + (size_t)e * FFN * HID, hb);
    moe_gemm2<<<dim3(HID / 64, NTOK / 128), 256, 0, stream>>>(
        hb, w2 + (size_t)e * HID * FFN, cw, e, out);
  }
}
